// MetaLADecoderLayer_39986145525846
// MI455X (gfx1250) — matrix operand order repaired and checked by reading, not compiled, not run on hardware
//
#include <hip/hip_runtime.h>
#include <hip/hip_bf16.h>
#include <math.h>

#define LL   2048
#define BB   4
#define DDim 1024
#define HH   16
#define DKK  512
#define GLUD 2816
#define MM   (LL * BB)

typedef __attribute__((ext_vector_type(16))) __bf16 v16bf;
typedef __attribute__((ext_vector_type(8)))  float  v8f;

// ---------------------------------------------------------------- helpers
__device__ __forceinline__ unsigned short f2bf(float f) {
    union { float f; unsigned int u; } c; c.f = f;
    unsigned int r = c.u + 0x7FFFu + ((c.u >> 16) & 1u);   // round-to-nearest-even
    return (unsigned short)(r >> 16);
}
__device__ __forceinline__ float bf2f(unsigned short h) {
    union { unsigned int u; float f; } c; c.u = ((unsigned int)h) << 16; return c.f;
}
__device__ __forceinline__ float siluf(float x) { return x / (1.f + __expf(-x)); }
__device__ __forceinline__ float sigmf(float x) { return 1.f / (1.f + __expf(-x)); }

// ---------------------------------------------------------------- f32 -> bf16 weight convert
__global__ __launch_bounds__(256) void cvt_bf16_kernel(const float* __restrict__ s,
                                                       unsigned short* __restrict__ d, int n) {
    int i = blockIdx.x * 256 + threadIdx.x;
    if (i < n) d[i] = f2bf(s[i]);
}

// ---------------------------------------------------------------- LN over D, (L,B,D) -> (B,L,D) f32
__global__ __launch_bounds__(256) void ln_transpose_kernel(const float* __restrict__ x,
                                                           const float* __restrict__ w,
                                                           const float* __restrict__ b,
                                                           float* __restrict__ out) {
    const int token = blockIdx.x;              // (l,b) row of x
    const int l = token >> 2, bb = token & 3;
    const float* xr = x + (size_t)token * DDim;
    const int tid = threadIdx.x;
    float s = 0.f, s2 = 0.f;
    for (int i = tid; i < DDim; i += 256) { float v = xr[i]; s += v; s2 += v * v; }
    __shared__ float red[2][8];
    for (int off = 16; off > 0; off >>= 1) { s += __shfl_down(s, off); s2 += __shfl_down(s2, off); }
    if ((tid & 31) == 0) { red[0][tid >> 5] = s; red[1][tid >> 5] = s2; }
    __syncthreads();
    if (tid == 0) {
        float t = 0.f, t2 = 0.f;
        for (int i = 0; i < 8; ++i) { t += red[0][i]; t2 += red[1][i]; }
        red[0][0] = t; red[1][0] = t2;
    }
    __syncthreads();
    const float mean = red[0][0] * (1.f / DDim);
    const float var  = red[1][0] * (1.f / DDim) - mean * mean;
    const float inv  = rsqrtf(var + 1e-5f);
    float* orow = out + ((size_t)bb * LL + l) * DDim;
    for (int i = tid; i < DDim; i += 256)
        orow[i] = (xr[i] - mean) * inv * w[i] + b[i];
}

// ---------------------------------------------------------------- LN over D, (L,B,D) -> bf16 same order
__global__ __launch_bounds__(256) void ln_bf16_kernel(const float* __restrict__ x,
                                                      const float* __restrict__ w,
                                                      const float* __restrict__ b,
                                                      unsigned short* __restrict__ out) {
    const int token = blockIdx.x;
    const float* xr = x + (size_t)token * DDim;
    const int tid = threadIdx.x;
    float s = 0.f, s2 = 0.f;
    for (int i = tid; i < DDim; i += 256) { float v = xr[i]; s += v; s2 += v * v; }
    __shared__ float red[2][8];
    for (int off = 16; off > 0; off >>= 1) { s += __shfl_down(s, off); s2 += __shfl_down(s2, off); }
    if ((tid & 31) == 0) { red[0][tid >> 5] = s; red[1][tid >> 5] = s2; }
    __syncthreads();
    if (tid == 0) {
        float t = 0.f, t2 = 0.f;
        for (int i = 0; i < 8; ++i) { t += red[0][i]; t2 += red[1][i]; }
        red[0][0] = t; red[1][0] = t2;
    }
    __syncthreads();
    const float mean = red[0][0] * (1.f / DDim);
    const float var  = red[1][0] * (1.f / DDim) - mean * mean;
    const float inv  = rsqrtf(var + 1e-5f);
    unsigned short* orow = out + (size_t)token * DDim;
    for (int i = tid; i < DDim; i += 256)
        orow[i] = f2bf((xr[i] - mean) * inv * w[i] + b[i]);
}

// ---------------------------------------------------------------- causal depthwise conv (k=4) + SiLU -> bf16
__global__ __launch_bounds__(256) void conv_silu_kernel(const float* __restrict__ hln,   // (B,L,D)
                                                        const float* __restrict__ cw,    // (D,4)
                                                        unsigned short* __restrict__ ybf) {
    const int token = blockIdx.x;               // b*L + l
    const int b = token >> 11, l = token & (LL - 1);
    const size_t base = (size_t)token * DDim;
    for (int d = threadIdx.x; d < DDim; d += 256) {
        float acc = 0.f;
        #pragma unroll
        for (int i = 0; i < 4; ++i) {
            const int li = l - 3 + i;
            if (li >= 0) acc += hln[((size_t)b * LL + li) * DDim + d] * cw[d * 4 + i];
        }
        ybf[base + d] = f2bf(siluf(acc));
    }
}

// ---------------------------------------------------------------- gates: gk = log_sigmoid(kg)/16, k = 1-exp(gk)
__global__ __launch_bounds__(256) void gate_kernel(float* __restrict__ kg,
                                                   float* __restrict__ gk, int n) {
    int i = blockIdx.x * 256 + threadIdx.x;
    if (i >= n) return;
    float x  = kg[i];
    float ls = fminf(x, 0.f) - log1pf(__expf(-fabsf(x)));   // stable log-sigmoid
    float g  = ls * (1.f / 16.f);
    gk[i] = g;
    kg[i] = 1.f - __expf(g);
}

// ---------------------------------------------------------------- GLA recurrent scan, one block per (b,h)
// thread t: vv = t>>2 (dv index), owns 8 consecutive dk entries of S[:,vv]
__global__ __launch_bounds__(256) void gla_scan_kernel(const float* __restrict__ q,   // (B,L,DK)
                                                       const float* __restrict__ k,
                                                       const float* __restrict__ gk,
                                                       const float* __restrict__ v,   // (B,L,D)
                                                       const float* __restrict__ aug, // (DK,)
                                                       float* __restrict__ o) {       // (B,L,D)
    const int bh = blockIdx.x;
    const int b = bh >> 4, h = bh & 15;
    const int tid = threadIdx.x;
    const int vv = tid >> 2, sub = tid & 3, kk0 = sub * 8;

    float S[8];
    #pragma unroll
    for (int j = 0; j < 8; ++j) S[j] = 0.f;
    float augv[8];
    #pragma unroll
    for (int j = 0; j < 8; ++j) augv[j] = aug[h * 32 + kk0 + j];

    const size_t qBase = (size_t)b * LL * DKK + h * 32 + kk0;
    const size_t vBase = (size_t)b * LL * DDim + h * 64 + vv;

    for (int l = 0; l < LL; ++l) {
        const float* qp = q  + qBase + (size_t)l * DKK;
        const float* kp = k  + qBase + (size_t)l * DKK;
        const float* gp = gk + qBase + (size_t)l * DKK;
        const float  vt = v[vBase + (size_t)l * DDim];
        float p = 0.f, pa = 0.f;
        #pragma unroll
        for (int j = 0; j < 8; ++j) {
            const float qj = qp[j], kj = kp[j], gj = gp[j];
            S[j] = __expf(gj) * S[j] + kj * vt;
            p  += qj * S[j];
            pa += qj * augv[j] * kj;
        }
        // reduce across the 4 lanes sharing this vv (wave32 shuffles)
        p  += __shfl_xor(p, 1);  p  += __shfl_xor(p, 2);
        pa += __shfl_xor(pa, 1); pa += __shfl_xor(pa, 2);
        const float ot = p + sigmf(pa * vt);
        if (sub == 0) o[((size_t)b * LL + l) * DDim + h * 64 + vv] = ot;
    }
}

// ---------------------------------------------------------------- per-head group-norm + silu(g) gate -> bf16
__global__ __launch_bounds__(256) void gn_gate_kernel(const float* __restrict__ o,
                                                      const float* __restrict__ g,
                                                      unsigned short* __restrict__ obf) {
    const int token = blockIdx.x;               // (B,L)
    const int tid = threadIdx.x;
    const int h = tid >> 4, j0 = (tid & 15) * 4;     // 16 threads per head, 4 elems each
    const size_t base = (size_t)token * DDim + h * 64 + j0;
    float xv[4];
    #pragma unroll
    for (int t = 0; t < 4; ++t) xv[t] = o[base + t];
    float s = xv[0] + xv[1] + xv[2] + xv[3];
    float s2 = xv[0] * xv[0] + xv[1] * xv[1] + xv[2] * xv[2] + xv[3] * xv[3];
    #pragma unroll
    for (int m = 1; m < 16; m <<= 1) { s += __shfl_xor(s, m); s2 += __shfl_xor(s2, m); }
    const float mean = s * (1.f / 64.f);
    const float var  = s2 * (1.f / 64.f) - mean * mean;
    const float inv  = rsqrtf(var + 1e-5f);
    #pragma unroll
    for (int t = 0; t < 4; ++t) {
        const float gg = g[base + t];
        obf[base + t] = f2bf((xv[t] - mean) * inv * siluf(gg));
    }
}

// ---------------------------------------------------------------- WMMA bf16 GEMM with fused epilogues
enum { EPI_NONE = 0, EPI_BIAS = 1, EPI_SILU_BF = 2, EPI_MUL_BF = 3, EPI_RESID_T = 4, EPI_ADD = 5 };

template <int EPI>
__global__ __launch_bounds__(256) void gemm_bf16_wmma(
    const unsigned short* __restrict__ A,     // (M,K) bf16 row-major
    const unsigned short* __restrict__ Bw,    // (K,N) bf16 row-major
    int M, int N, int K,
    float* __restrict__ outF,
    unsigned short* __restrict__ outBF,
    const float* __restrict__ bias,
    const float* __restrict__ addsrc,
    const unsigned short* __restrict__ mulsrc) {
    __shared__ __align__(16) unsigned short As[128][40];   // [m][k], 80B row stride (16B aligned)
    __shared__ __align__(16) unsigned short Bs[128][40];   // transposed: [n][k]

    const int tid  = threadIdx.x;
    const int m0   = blockIdx.y * 128;
    const int n0   = blockIdx.x * 128;
    const int wave = tid >> 5, lane = tid & 31;
    const int waveM = (wave & 1) * 64;        // 2 waves in M
    const int waveN = (wave >> 1) * 32;       // 4 waves in N
    const int l16 = lane & 15;
    const int hiA = (lane >> 4) * 8;          // A/K sub-offset and C/M sub-offset (ISA 7.12.2)
    const int hiB = (lane >> 4) * 16;         // unused; B fragments take their K sub-offset from hiA

    v8f acc[4][2] = {};

    const int aRow = tid >> 1, aCol = (tid & 1) * 16;
    const int bK   = tid >> 3, bN   = (tid & 7) * 16;

    for (int kt = 0; kt < K; kt += 32) {
        // ---- stage A tile (128x32) and B tile (32x128, transposed) into LDS
        const uint4* ag = (const uint4*)(A + (size_t)(m0 + aRow) * K + kt + aCol);
        const uint4 a0 = ag[0], a1 = ag[1];
        const uint4* bg = (const uint4*)(Bw + (size_t)(kt + bK) * N + n0 + bN);
        const uint4 b0 = bg[0], b1 = bg[1];
        if (kt + 32 < K) {                            // gfx1250 global_prefetch_b8
            __builtin_prefetch(A + (size_t)(m0 + aRow) * K + kt + 32 + aCol, 0, 0);
            __builtin_prefetch(Bw + (size_t)(kt + 32 + bK) * N + n0 + bN, 0, 0);
        }
        *(uint4*)&As[aRow][aCol]     = a0;
        *(uint4*)&As[aRow][aCol + 8] = a1;
        unsigned short tmp[16];
        *(uint4*)&tmp[0] = b0;
        *(uint4*)&tmp[8] = b1;
        #pragma unroll
        for (int j = 0; j < 16; ++j) Bs[bN + j][bK] = tmp[j];
        __syncthreads();

        // ---- build fragments per ISA lane layouts and issue WMMAs
        union Frag { unsigned int u[8]; v16bf v; };
        Frag bf[2];
        #pragma unroll
        for (int nt = 0; nt < 2; ++nt) {
            const int n = waveN + nt * 16 + l16;
            #pragma unroll
            for (int p = 0; p < 8; ++p)
                bf[nt].u[p] = *(const unsigned int*)&Bs[n][2 * (p & 3) + (p >> 2) * 16 + hiA];
        }
        #pragma unroll
        for (int mt = 0; mt < 4; ++mt) {
            const int m = waveM + mt * 16 + l16;
            Frag af;
            #pragma unroll
            for (int p = 0; p < 4; ++p)
                af.u[p] = *(const unsigned int*)&As[m][2 * p + hiA];
            #pragma unroll
            for (int p = 0; p < 4; ++p)
                af.u[4 + p] = *(const unsigned int*)&As[m][16 + 2 * p + hiA];
            #pragma unroll
            for (int nt = 0; nt < 2; ++nt)
                acc[mt][nt] = __builtin_amdgcn_wmma_f32_16x16x32_bf16(
                    false, af.v, false, bf[nt].v, (short)0, acc[mt][nt], false, false);
        }
        __syncthreads();
    }

    // ---- epilogue (C/D layout: VGPR r -> M = r + 8*(lane>=16), N = lane&15)
    #pragma unroll
    for (int mt = 0; mt < 4; ++mt) {
        #pragma unroll
        for (int nt = 0; nt < 2; ++nt) {
            #pragma unroll
            for (int r = 0; r < 8; ++r) {
                const int gm = m0 + waveM + mt * 16 + r + hiA;
                const int gn = n0 + waveN + nt * 16 + l16;
                const float val = acc[mt][nt][r];
                const size_t idx = (size_t)gm * N + gn;
                if (EPI == EPI_NONE) {
                    outF[idx] = val;
                } else if (EPI == EPI_BIAS) {
                    outF[idx] = val + bias[gn];
                } else if (EPI == EPI_SILU_BF) {
                    outBF[idx] = f2bf(siluf(val));
                } else if (EPI == EPI_MUL_BF) {
                    outBF[idx] = f2bf(bf2f(mulsrc[idx]) * val);
                } else if (EPI == EPI_RESID_T) {       // (B,L) row -> (L,B) row + residual
                    const int l = gm & (LL - 1);
                    const int b = gm >> 11;
                    const size_t didx = ((size_t)l * BB + b) * N + gn;
                    outF[didx] = val + addsrc[didx];
                } else {                               // EPI_ADD
                    outF[idx] = val + addsrc[idx];
                }
            }
        }
    }
}

// ---------------------------------------------------------------- launcher
extern "C" void kernel_launch(void* const* d_in, const int* in_sizes, int n_in,
                              void* d_out, int out_size, void* d_ws, size_t ws_size,
                              hipStream_t stream) {
    const float* x      = (const float*)d_in[0];
    const float* tn_w   = (const float*)d_in[1];
    const float* tn_b   = (const float*)d_in[2];
    const float* conv_w = (const float*)d_in[3];
    const float* q_w    = (const float*)d_in[4];
    const float* kg_w   = (const float*)d_in[5];
    const float* v_w    = (const float*)d_in[6];
    const float* g_w    = (const float*)d_in[7];
    const float* g_b    = (const float*)d_in[8];
    const float* out_w  = (const float*)d_in[9];
    const float* aug    = (const float*)d_in[10];
    const float* cn_w   = (const float*)d_in[11];
    const float* cn_b   = (const float*)d_in[12];
    const float* l1_w   = (const float*)d_in[13];
    const float* l2_w   = (const float*)d_in[14];
    const float* l3_w   = (const float*)d_in[15];

    char* ws = (char*)d_ws;
    size_t off = 0;
    auto alloc = [&](size_t bytes) -> char* {
        char* p = ws + off;
        off += (bytes + 255) & ~(size_t)255;
        return p;
    };

    // persistent bf16 weights
    unsigned short* w_q   = (unsigned short*)alloc((size_t)DDim * DKK * 2);
    unsigned short* w_kg  = (unsigned short*)alloc((size_t)DDim * DKK * 2);
    unsigned short* w_v   = (unsigned short*)alloc((size_t)DDim * DDim * 2);
    unsigned short* w_g   = (unsigned short*)alloc((size_t)DDim * DDim * 2);
    unsigned short* w_out = (unsigned short*)alloc((size_t)DDim * DDim * 2);
    unsigned short* w_l1  = (unsigned short*)alloc((size_t)DDim * GLUD * 2);
    unsigned short* w_l2  = (unsigned short*)alloc((size_t)DDim * GLUD * 2);
    unsigned short* w_l3  = (unsigned short*)alloc((size_t)GLUD * DDim * 2);

    const size_t reuse_mark = off;              // attention-phase region, reused for GLU
    unsigned short* ybf  = (unsigned short*)alloc((size_t)MM * DDim * 2);
    float* qbuf  = (float*)alloc((size_t)MM * DKK * 4);
    float* kbuf  = (float*)alloc((size_t)MM * DKK * 4);
    float* gkbuf = (float*)alloc((size_t)MM * DKK * 4);
    float* vbuf  = (float*)alloc((size_t)MM * DDim * 4);   // doubles as hln before v GEMM
    float* gbuf  = (float*)alloc((size_t)MM * DDim * 4);
    float* obuf  = (float*)alloc((size_t)MM * DDim * 4);
    unsigned short* obf = (unsigned short*)alloc((size_t)MM * DDim * 2);
    float* x1    = (float*)alloc((size_t)MM * DDim * 4);
    unsigned short* zbf = (unsigned short*)alloc((size_t)MM * DDim * 2);
    // GLU intermediates (2 x 46MB) overlap the 184MB freed region [ybf .. obf]
    unsigned short* abf   = (unsigned short*)(ws + reuse_mark);
    unsigned short* glubf = abf + (size_t)MM * GLUD;

    float* outp = (float*)d_out;

    // 1) weights -> bf16
    auto cvt = [&](const float* s, unsigned short* d, int n) {
        cvt_bf16_kernel<<<(n + 255) / 256, 256, 0, stream>>>(s, d, n);
    };
    cvt(q_w,   w_q,   DDim * DKK);
    cvt(kg_w,  w_kg,  DDim * DKK);
    cvt(v_w,   w_v,   DDim * DDim);
    cvt(g_w,   w_g,   DDim * DDim);
    cvt(out_w, w_out, DDim * DDim);
    cvt(l1_w,  w_l1,  DDim * GLUD);
    cvt(l2_w,  w_l2,  DDim * GLUD);
    cvt(l3_w,  w_l3,  GLUD * DDim);

    // 2) LN + transpose -> hln (B,L,D); 3) conv + SiLU -> ybf
    ln_transpose_kernel<<<MM, 256, 0, stream>>>(x, tn_w, tn_b, vbuf /*hln*/);
    conv_silu_kernel<<<MM, 256, 0, stream>>>(vbuf, conv_w, ybf);

    // 4) projection GEMMs (WMMA bf16)
    const dim3 gq(DKK / 128, MM / 128), gd(DDim / 128, MM / 128), gglu(GLUD / 128, MM / 128);
    gemm_bf16_wmma<EPI_NONE><<<gq, 256, 0, stream>>>(ybf, w_q,  MM, DKK,  DDim, qbuf,  nullptr, nullptr, nullptr, nullptr);
    gemm_bf16_wmma<EPI_NONE><<<gq, 256, 0, stream>>>(ybf, w_kg, MM, DKK,  DDim, kbuf,  nullptr, nullptr, nullptr, nullptr);
    gemm_bf16_wmma<EPI_NONE><<<gd, 256, 0, stream>>>(ybf, w_v,  MM, DDim, DDim, vbuf,  nullptr, nullptr, nullptr, nullptr);
    gemm_bf16_wmma<EPI_BIAS><<<gd, 256, 0, stream>>>(ybf, w_g,  MM, DDim, DDim, gbuf,  nullptr, g_b,     nullptr, nullptr);

    // 5) gates; 6) recurrent GLA scan; 7) group-norm + silu(g) gate -> bf16
    gate_kernel<<<(MM * DKK + 255) / 256, 256, 0, stream>>>(kbuf, gkbuf, MM * DKK);
    gla_scan_kernel<<<BB * HH, 256, 0, stream>>>(qbuf, kbuf, gkbuf, vbuf, aug, obuf);
    gn_gate_kernel<<<MM, 256, 0, stream>>>(obuf, gbuf, obf);

    // 8) out projection + transpose + residual -> x1 (L,B,D)
    gemm_bf16_wmma<EPI_RESID_T><<<gd, 256, 0, stream>>>(obf, w_out, MM, DDim, DDim, x1, nullptr, nullptr, x, nullptr);

    // 9) second LN -> zbf; 10) GLU MLP with fused silu/mul/add epilogues
    ln_bf16_kernel<<<MM, 256, 0, stream>>>(x1, cn_w, cn_b, zbf);
    gemm_bf16_wmma<EPI_SILU_BF><<<gglu, 256, 0, stream>>>(zbf, w_l1, MM, GLUD, DDim, nullptr, abf,   nullptr, nullptr, nullptr);
    gemm_bf16_wmma<EPI_MUL_BF><<<gglu, 256, 0, stream>>>(zbf, w_l2, MM, GLUD, DDim, nullptr, glubf, nullptr, nullptr, abf);
    gemm_bf16_wmma<EPI_ADD><<<gd, 256, 0, stream>>>(glubf, w_l3, MM, DDim, GLUD, outp, nullptr, nullptr, x1, nullptr);
}
